// NPILSTM_80058190397590
// MI455X (gfx1250) — matrix operand order repaired and checked by reading, not compiled, not run on hardware
//
#include <hip/hip_runtime.h>

// ---------------------------------------------------------------------------
// LSTM (B=64, SEQ=128, E=512, H=1024) on gfx1250 via v_wmma_f32_16x16x32_bf16
// ---------------------------------------------------------------------------

typedef __attribute__((ext_vector_type(16))) __bf16 v16bf;
typedef __attribute__((ext_vector_type(8)))  __bf16 v8bf;
typedef __attribute__((ext_vector_type(8)))  float  v8f;

#define BATCH   64
#define SEQLEN  128
#define EMBED   512
#define HDIM    1024
#define STEPS   127      // seq - 1
#define T_START 65       // num_unary + 2 = (128-2)/2 + 2
#define OUT_S   62       // 127 - 65

static __device__ __forceinline__ unsigned short f2bf(float f) {
    unsigned int u = __float_as_uint(f);
    u += 0x7FFFu + ((u >> 16) & 1u);          // round-to-nearest-even
    return (unsigned short)(u >> 16);
}

static __device__ __forceinline__ float sigmoidf_(float x) {
    return 1.0f / (1.0f + __expf(-x));
}

// A fragment, 16x32 bf16, row-major source with leading dim `ld` (elements).
// lane<16: row = lane,    K in {koff..koff+15}, koff=0
// lane>=16: same row,     koff=16
static __device__ __forceinline__ v16bf load_a(const unsigned short* __restrict__ base,
                                               int ld, int row, int k0, int koff) {
    const unsigned short* p = base + (size_t)row * ld + k0 + koff;
    v8bf lo = *(const v8bf*)(const void*)(p);
    v8bf hi = *(const v8bf*)(const void*)(p + 8);
    return __builtin_shufflevector(lo, hi, 0,1,2,3,4,5,6,7,8,9,10,11,12,13,14,15);
}

// -------------------------- prep kernels -----------------------------------

__global__ void k_cvt_bf16(const float* __restrict__ src, unsigned short* __restrict__ dst, int n) {
    int i = blockIdx.x * blockDim.x + threadIdx.x;
    if (i < n) dst[i] = f2bf(src[i]);
}

__global__ void k_bias_combine(const float* __restrict__ b_ih, const float* __restrict__ b_hh,
                               float* __restrict__ bias) {
    int i = blockIdx.x * blockDim.x + threadIdx.x;
    if (i < 4 * HDIM) bias[i] = b_ih[i] + b_hh[i];
}

// Gather embeddings for all timesteps 0..STEPS-1 into bf16 x[t][b][e]
__global__ void k_embed(const int* __restrict__ prob, const float* __restrict__ table,
                        unsigned short* __restrict__ x_bf) {
    int tb = blockIdx.x;           // t * BATCH + b
    int t  = tb / BATCH;
    int b  = tb % BATCH;
    int tok = prob[b * SEQLEN + t];
    const float* row = table + (size_t)tok * EMBED;
    unsigned short* out = x_bf + (size_t)tb * EMBED;
    for (int k = threadIdx.x; k < EMBED; k += blockDim.x) out[k] = f2bf(row[k]);
}

// -------------------------- LSTM step (WMMA) -------------------------------
// grid = 64 blocks x 128 threads (4 waves). 256 waves total.
// wave id -> (m_tile 0..3, n_tile 0..63); each wave computes the 16x16
// (batch, j) patch for ALL FOUR gates and performs the cell update in-place.

__global__ __launch_bounds__(128) void k_lstm_step(
    const unsigned short* __restrict__ x_bf,   // [64][512]  this timestep
    const unsigned short* __restrict__ h_in,   // [64][1024] bf16
    const unsigned short* __restrict__ w_ih,   // [4096][512] bf16
    const unsigned short* __restrict__ w_hh,   // [4096][1024] bf16
    const float*          __restrict__ bias,   // [4096] = b_ih + b_hh
    float*                __restrict__ c,      // [64][1024] f32 (in/out, element-owned)
    unsigned short*       __restrict__ h_out,  // [64][1024] bf16
    float*                __restrict__ hs)     // nullptr or [64][1024] f32 slot
{
    const int lane   = threadIdx.x & 31;
    const int wid    = blockIdx.x * (blockDim.x >> 5) + (threadIdx.x >> 5);
    const int m_tile = wid >> 6;        // 0..3  (batch / 16)
    const int n_tile = wid & 63;        // 0..63 (j / 16)

    const int rowA   = m_tile * 16 + (lane & 15);
    const int koff_a = (lane >> 4) * 16;
    const int koff_b = (lane >> 4) * 16;
    const int colB   = lane & 15;
    const int n0     = n_tile * 16;

    v8f acc[4] = {};   // i, f, g, o accumulators

    // ---- gates += x @ w_ih^T  (K = 512) ----
    #pragma unroll 4
    for (int k0 = 0; k0 < EMBED; k0 += 32) {
        v16bf a = load_a(x_bf, EMBED, rowA, k0, koff_a);
        #pragma unroll
        for (int q = 0; q < 4; ++q) {
            const unsigned short* wr = w_ih + ((size_t)(q * HDIM + n0 + colB)) * EMBED;
            v16bf bfrag = *(const v16bf*)(const void*)(wr + k0 + koff_b);
            acc[q] = __builtin_amdgcn_wmma_f32_16x16x32_bf16(
                         false, a, false, bfrag, (short)0, acc[q], false, false);
        }
    }

    // ---- gates += h @ w_hh^T  (K = 1024) ----
    #pragma unroll 4
    for (int k0 = 0; k0 < HDIM; k0 += 32) {
        v16bf a = load_a(h_in, HDIM, rowA, k0, koff_a);
        #pragma unroll
        for (int q = 0; q < 4; ++q) {
            const unsigned short* wr = w_hh + ((size_t)(q * HDIM + n0 + colB)) * HDIM;
            v16bf bfrag = *(const v16bf*)(const void*)(wr + k0 + koff_b);
            acc[q] = __builtin_amdgcn_wmma_f32_16x16x32_bf16(
                         false, a, false, bfrag, (short)0, acc[q], false, false);
        }
    }

    // ---- cell update (each lane owns 8 elements: 8 batch rows, 1 column) ----
    const int j     = n0 + colB;
    const int rbase = (lane < 16) ? 0 : 8;   // C/D layout: lanes>=16 hold M=8..15
    #pragma unroll
    for (int r = 0; r < 8; ++r) {
        int b = m_tile * 16 + rbase + r;
        float iv = acc[0][r] + bias[j];
        float fv = acc[1][r] + bias[HDIM + j];
        float gv = acc[2][r] + bias[2 * HDIM + j];
        float ov = acc[3][r] + bias[3 * HDIM + j];
        float is = sigmoidf_(iv);
        float fs = sigmoidf_(fv);
        float gt = tanhf(gv);
        float os = sigmoidf_(ov);
        size_t idx = (size_t)b * HDIM + j;
        float cn = fs * c[idx] + is * gt;
        float hn = os * tanhf(cn);
        c[idx]     = cn;
        h_out[idx] = f2bf(hn);
        if (hs) hs[idx] = hn;
    }
}

// -------------------------- final projection -------------------------------
// out[b][s] = dot(hs[s][b][:], w_ans) + b_ans    (deterministic tree reduce)

__global__ __launch_bounds__(256) void k_pred(const float* __restrict__ hs,
                                              const float* __restrict__ w_ans,
                                              const float* __restrict__ b_ans,
                                              float* __restrict__ out) {
    int s = blockIdx.x / BATCH;
    int b = blockIdx.x % BATCH;
    const float* h = hs + ((size_t)s * BATCH + b) * HDIM;
    float sum = 0.0f;
    for (int k = threadIdx.x; k < HDIM; k += 256) sum += h[k] * w_ans[k];
    #pragma unroll
    for (int off = 16; off > 0; off >>= 1) sum += __shfl_down(sum, off, 32);
    __shared__ float red[8];
    if ((threadIdx.x & 31) == 0) red[threadIdx.x >> 5] = sum;
    __syncthreads();
    if (threadIdx.x == 0) {
        float tot = 0.0f;
        #pragma unroll
        for (int w = 0; w < 8; ++w) tot += red[w];
        out[b * OUT_S + s] = tot + b_ans[0];
    }
}

// ---------------------------------------------------------------------------

extern "C" void kernel_launch(void* const* d_in, const int* in_sizes, int n_in,
                              void* d_out, int out_size, void* d_ws, size_t ws_size,
                              hipStream_t stream) {
    (void)in_sizes; (void)n_in; (void)out_size; (void)ws_size;

    const int*   prob  = (const int*)  d_in[0];
    /* d_in[1] = ans (unused) */
    const float* table = (const float*)d_in[2];
    const float* w_ih  = (const float*)d_in[3];
    const float* w_hh  = (const float*)d_in[4];
    const float* b_ih  = (const float*)d_in[5];
    const float* b_hh  = (const float*)d_in[6];
    const float* w_ans = (const float*)d_in[7];
    const float* b_ans = (const float*)d_in[8];
    const float* h0    = (const float*)d_in[9];
    const float* c0    = (const float*)d_in[10];

    char*  ws  = (char*)d_ws;
    size_t off = 0;
    auto alloc = [&](size_t bytes) -> void* {
        void* p = ws + off;
        off = (off + bytes + 255) & ~(size_t)255;
        return p;
    };

    unsigned short* w_ih_bf = (unsigned short*)alloc((size_t)4 * HDIM * EMBED * 2);
    unsigned short* w_hh_bf = (unsigned short*)alloc((size_t)4 * HDIM * HDIM  * 2);
    unsigned short* x_bf    = (unsigned short*)alloc((size_t)STEPS * BATCH * EMBED * 2);
    unsigned short* h_bf0   = (unsigned short*)alloc((size_t)BATCH * HDIM * 2);
    unsigned short* h_bf1   = (unsigned short*)alloc((size_t)BATCH * HDIM * 2);
    float*          c_buf   = (float*)         alloc((size_t)BATCH * HDIM * 4);
    float*          bias    = (float*)         alloc((size_t)4 * HDIM * 4);
    float*          hs_all  = (float*)         alloc((size_t)OUT_S * BATCH * HDIM * 4);

    // --- prep ---
    {
        int n = 4 * HDIM * EMBED;
        k_cvt_bf16<<<(n + 255) / 256, 256, 0, stream>>>(w_ih, w_ih_bf, n);
    }
    {
        int n = 4 * HDIM * HDIM;
        k_cvt_bf16<<<(n + 255) / 256, 256, 0, stream>>>(w_hh, w_hh_bf, n);
    }
    {
        int n = BATCH * HDIM;
        k_cvt_bf16<<<(n + 255) / 256, 256, 0, stream>>>(h0, h_bf0, n);
    }
    hipMemcpyAsync(c_buf, c0, (size_t)BATCH * HDIM * sizeof(float),
                   hipMemcpyDeviceToDevice, stream);
    k_bias_combine<<<(4 * HDIM + 255) / 256, 256, 0, stream>>>(b_ih, b_hh, bias);
    k_embed<<<STEPS * BATCH, 128, 0, stream>>>(prob, table, x_bf);

    // --- recurrence: one WMMA kernel per timestep ---
    for (int t = 0; t < STEPS; ++t) {
        const unsigned short* hin  = (t & 1) ? h_bf1 : h_bf0;
        unsigned short*       hout = (t & 1) ? h_bf0 : h_bf1;
        float* hs_slot = (t >= T_START)
                       ? (hs_all + (size_t)(t - T_START) * BATCH * HDIM)
                       : nullptr;
        k_lstm_step<<<64, 128, 0, stream>>>(
            x_bf + (size_t)t * BATCH * EMBED,
            hin, w_ih_bf, w_hh_bf, bias, c_buf, hout, hs_slot);
    }

    // --- output projection ---
    k_pred<<<OUT_S * BATCH, 256, 0, stream>>>(hs_all, w_ans, b_ans, (float*)d_out);
}
